// GemmARLayer_50672024158952
// MI455X (gfx1250) — matrix operand order repaired and checked by reading, not compiled, not run on hardware
//
#include <hip/hip_runtime.h>

// GEMM-with-fused-all-reduce: C[M,N] = sum_r A[r,:,:] @ B[r,:,:]^T
// A: [8, 4096, 1792] fp32, B: [8, 4096, 1792] fp32, C: [4096, 4096] fp32.
// Compute-bound (AI ~900 FLOP/B) -> bf16x3 split-precision WMMA path.
//
// Fast path (needs 448 MiB workspace):
//   1) convert_planes: fp32 -> bf16 hi/lo planes, re-tiled to chunked layout
//      plane[r*56+c][row][32] so every tile is contiguous.
//   2) gemm_planes: 256x128 block tile, 64x64 wave tile (ds:wmma = 0.67),
//      async global->LDS copies (ASYNCcnt) + WMMA; no conversion VALU.
// Fallback (small ws): fused in-kernel conversion.

#define W_RANKS 8
#define M_DIM   4096
#define N_DIM   4096
#define K_DIM   1792

#define BLK_K 32
#define KCHUNKS (K_DIM / BLK_K)          // 56
#define TOTAL_T (W_RANKS * KCHUNKS)      // 448
#define LDSB  80   // 64B of bf16 row data + 16B pad: conflict-free ds_load_b128

// Planes kernel block tile
#define PBLK_M 256
#define PBLK_N 128
// Fused fallback block tile
#define BLK_M 128
#define BLK_N 128

// Plane geometry: [TOTAL_T][4096][32] bf16
#define PLANE_ELEMS ((size_t)TOTAL_T * 4096 * 32)   // 58,720,256
#define PLANE_BYTES (PLANE_ELEMS * 2)               // 117,440,512

typedef __attribute__((ext_vector_type(16))) __bf16   v16bf;
typedef __attribute__((ext_vector_type(8)))  __bf16   v8bf;
typedef __attribute__((ext_vector_type(2)))  __bf16   v2bf;
typedef __attribute__((ext_vector_type(8)))  float    v8f;
typedef __attribute__((ext_vector_type(4)))  float    f4;
typedef __attribute__((ext_vector_type(4)))  unsigned ui4;

union FragBF {
  v16bf v;
  v8bf  h[2];
};

__device__ __forceinline__ unsigned bf16_rne(float x) {
  unsigned u = __float_as_uint(x);
  u += 0x7FFFu + ((u >> 16) & 1u);      // round-to-nearest-even to bf16
  return u >> 16;
}

// Split two fp32 into packed bf16 hi pair + bf16 lo pair (x = hi + lo).
__device__ __forceinline__ void cvt_split_pair(float x0, float x1,
                                               unsigned& hi, unsigned& lo) {
#if __has_builtin(__builtin_amdgcn_cvt_pk_bf16_f32)
  union { v2bf v; unsigned u; } h, l;
  h.v = __builtin_amdgcn_cvt_pk_bf16_f32(x0, x1);
  float h0 = (float)h.v[0];
  float h1 = (float)h.v[1];
  l.v = __builtin_amdgcn_cvt_pk_bf16_f32(x0 - h0, x1 - h1);
  hi = h.u;
  lo = l.u;
#else
  unsigned h0 = bf16_rne(x0), h1 = bf16_rne(x1);
  hi = h0 | (h1 << 16);
  unsigned l0 = bf16_rne(x0 - __uint_as_float(h0 << 16));
  unsigned l1 = bf16_rne(x1 - __uint_as_float(h1 << 16));
  lo = l0 | (l1 << 16);
#endif
}

// Async global->LDS: INST_OFFSET applies to both LDS and memory addresses.
#define ASYNC_COPY2(ldsaddr, gaddr)                                          \
  asm volatile("global_load_async_to_lds_b128 %0, %1, off\n\t"               \
               "global_load_async_to_lds_b128 %0, %1, off offset:16"         \
               :: "v"(ldsaddr), "v"(gaddr) : "memory")
#define ASYNC_COPY4(ldsaddr, gaddr)                                          \
  asm volatile("global_load_async_to_lds_b128 %0, %1, off\n\t"               \
               "global_load_async_to_lds_b128 %0, %1, off offset:16\n\t"     \
               "global_load_async_to_lds_b128 %0, %1, off offset:32\n\t"     \
               "global_load_async_to_lds_b128 %0, %1, off offset:48"         \
               :: "v"(ldsaddr), "v"(gaddr) : "memory")

// ---------------------------------------------------------------------------
// Pass 1: fp32 [8][4096][1792] -> bf16 hi/lo planes [448][4096][32] (chunked).
// One thread handles one (matrix, chunk, row): reads 128B, writes 64B+64B.
// ---------------------------------------------------------------------------
__global__ __launch_bounds__(256)
void convert_planes(const float* __restrict__ A, const float* __restrict__ B,
                    unsigned short* __restrict__ Ahi,
                    unsigned short* __restrict__ Alo,
                    unsigned short* __restrict__ Bhi,
                    unsigned short* __restrict__ Blo) {
  const unsigned perMat = (unsigned)TOTAL_T * 4096u;      // 1,835,008
  unsigned g = blockIdx.x * 256u + threadIdx.x;           // < 2*perMat
  const unsigned mat = g / perMat;
  const unsigned rem = g - mat * perMat;
  const unsigned rc  = rem >> 12;                         // 0..447
  const unsigned row = rem & 4095u;
  const unsigned r  = rc / KCHUNKS;
  const unsigned c  = rc - r * KCHUNKS;

  const float* src = (mat == 0 ? A : B) +
      ((size_t)r * 4096 + row) * (size_t)K_DIM + (size_t)c * BLK_K;
  unsigned short* dH = (mat == 0 ? Ahi : Bhi);
  unsigned short* dL = (mat == 0 ? Alo : Blo);
  const size_t dOff = ((size_t)rc * 4096 + row) * BLK_K;

  f4 v[8];
#pragma unroll
  for (int i = 0; i < 8; ++i) v[i] = *(const f4*)(src + i * 4);

  ui4 ph[4], pl[4];
#pragma unroll
  for (int i = 0; i < 8; ++i) {
#pragma unroll
    for (int j = 0; j < 4; j += 2) {
      const int e = (i * 4 + j) >> 1;   // packed-uint index 0..15
      unsigned h, l;
      cvt_split_pair(v[i][j], v[i][j + 1], h, l);
      ph[e >> 2][e & 3] = h;
      pl[e >> 2][e & 3] = l;
    }
  }
#pragma unroll
  for (int i = 0; i < 4; ++i) {
    *(ui4*)(dH + dOff + i * 8) = ph[i];
    *(ui4*)(dL + dOff + i * 8) = pl[i];
  }
}

// ---------------------------------------------------------------------------
// Pass 2: GEMM on pre-converted planes.
// Block 256x128, 8 waves as 4(M) x 2(N), each wave a 64x64 tile (4x4 WMMA
// sub-tiles). Per chunk per wave: 32 ds_load_b128, 48 WMMA.
// ---------------------------------------------------------------------------
__global__ __launch_bounds__(256)
void gemm_planes(const unsigned short* __restrict__ Ahi,
                 const unsigned short* __restrict__ Alo,
                 const unsigned short* __restrict__ Bhi,
                 const unsigned short* __restrict__ Blo,
                 float* __restrict__ C) {
  __shared__ __align__(16) unsigned char sAhi[2][PBLK_M * LDSB];
  __shared__ __align__(16) unsigned char sAlo[2][PBLK_M * LDSB];
  __shared__ __align__(16) unsigned char sBhi[2][PBLK_N * LDSB];
  __shared__ __align__(16) unsigned char sBlo[2][PBLK_N * LDSB];

  const int tid   = threadIdx.x;
  const int lane  = tid & 31;
  const int wave  = tid >> 5;
  const int waveM = wave >> 1;            // 0..3 -> 64-row stripe
  const int waveN = wave & 1;             // 0..1 -> 64-col stripe
  const int m0 = blockIdx.y * PBLK_M;
  const int n0 = blockIdx.x * PBLK_N;

  // Copy mapping:
  //  A tile (256 rows x 64B): thread t moves row t (one pair + 4 issues)
  //  B tile (128 rows x 64B): thread t moves row t/2, 32B-half t&1
  const unsigned aRowOff = (unsigned)tid * LDSB;
  const unsigned bRowOff =
      (unsigned)(tid >> 1) * LDSB + (unsigned)(tid & 1) * 32u;

  const int lr = lane & 15;
  const int lh = (lane >> 4) & 1;

  unsigned aFragOff[4], bFragOff[4];
#pragma unroll
  for (int mi = 0; mi < 4; ++mi)
    aFragOff[mi] =
        (unsigned)(waveM * 64 + mi * 16 + lr) * LDSB + (unsigned)lh * 16u;
#pragma unroll
  for (int ni = 0; ni < 4; ++ni)
    bFragOff[ni] =
        (unsigned)(waveN * 64 + ni * 16 + lr) * LDSB + (unsigned)lh * 16u;

  const v8f vzero = {0.f, 0.f, 0.f, 0.f, 0.f, 0.f, 0.f, 0.f};
  v8f acc[4][4];
#pragma unroll
  for (int mi = 0; mi < 4; ++mi)
#pragma unroll
    for (int ni = 0; ni < 4; ++ni) acc[mi][ni] = vzero;

  auto issue_copy = [&](int t, int buf) {
    const size_t aOff = ((size_t)t * 4096 + (size_t)(m0 + tid)) * BLK_K;
    const size_t bOff =
        ((size_t)t * 4096 + (size_t)(n0 + (tid >> 1))) * BLK_K +
        (size_t)(tid & 1) * 16;
    ASYNC_COPY4((unsigned)(size_t)(sAhi[buf] + aRowOff),
                (unsigned long long)(Ahi + aOff));
    ASYNC_COPY4((unsigned)(size_t)(sAlo[buf] + aRowOff),
                (unsigned long long)(Alo + aOff));
    ASYNC_COPY2((unsigned)(size_t)(sBhi[buf] + bRowOff),
                (unsigned long long)(Bhi + bOff));
    ASYNC_COPY2((unsigned)(size_t)(sBlo[buf] + bRowOff),
                (unsigned long long)(Blo + bOff));
  };

  auto compute_tile = [&](int buf) {
    // A fragments (reused across all ni) held live; B fragments per-ni.
    FragBF fahi[4], falo[4];
#pragma unroll
    for (int mi = 0; mi < 4; ++mi) {
      const unsigned off = aFragOff[mi];
      fahi[mi].h[0] = *(const v8bf*)(sAhi[buf] + off);
      fahi[mi].h[1] = *(const v8bf*)(sAhi[buf] + off + 32);
      falo[mi].h[0] = *(const v8bf*)(sAlo[buf] + off);
      falo[mi].h[1] = *(const v8bf*)(sAlo[buf] + off + 32);
    }
#pragma unroll
    for (int ni = 0; ni < 4; ++ni) {
      const unsigned off = bFragOff[ni];
      FragBF fbhi, fblo;
      fbhi.h[0] = *(const v8bf*)(sBhi[buf] + off);
      fbhi.h[1] = *(const v8bf*)(sBhi[buf] + off + 32);
      fblo.h[0] = *(const v8bf*)(sBlo[buf] + off);
      fblo.h[1] = *(const v8bf*)(sBlo[buf] + off + 32);
#pragma unroll
      for (int mi = 0; mi < 4; ++mi) {
        acc[mi][ni] = __builtin_amdgcn_wmma_f32_16x16x32_bf16(
            false, fahi[mi].v, false, fbhi.v, (short)0, acc[mi][ni],
            false, false);
        acc[mi][ni] = __builtin_amdgcn_wmma_f32_16x16x32_bf16(
            false, falo[mi].v, false, fbhi.v, (short)0, acc[mi][ni],
            false, false);
        acc[mi][ni] = __builtin_amdgcn_wmma_f32_16x16x32_bf16(
            false, fahi[mi].v, false, fblo.v, (short)0, acc[mi][ni],
            false, false);
      }
    }
  };

  issue_copy(0, 0);
  asm volatile("s_wait_asynccnt 0x0" ::: "memory");
  __syncthreads();

  for (int t = 0; t < TOTAL_T; ++t) {
    const int pb = t & 1;
    if (t + 1 < TOTAL_T) issue_copy(t + 1, pb ^ 1);  // fire-and-forget
    compute_tile(pb);
    asm volatile("s_wait_asynccnt 0x0" ::: "memory");
    __syncthreads();
  }

#pragma unroll
  for (int mi = 0; mi < 4; ++mi) {
#pragma unroll
    for (int ni = 0; ni < 4; ++ni) {
      const int row0 = m0 + waveM * 64 + mi * 16 + lh * 8;
      const int col  = n0 + waveN * 64 + ni * 16 + lr;
      float* cp = C + (size_t)row0 * N_DIM + col;
#pragma unroll
      for (int j = 0; j < 8; ++j) cp[(size_t)j * N_DIM] = acc[mi][ni][j];
    }
  }
}

// ---------------------------------------------------------------------------
// Fallback: fused in-kernel conversion (used when workspace is too small).
// ---------------------------------------------------------------------------
struct Tile {
  f4 a[4];
  f4 b[4];
};

__global__ __launch_bounds__(256)
void gemm_fused(const float* __restrict__ A, const float* __restrict__ B,
                float* __restrict__ C) {
  __shared__ __align__(16) unsigned char sAhi[2][BLK_M * LDSB];
  __shared__ __align__(16) unsigned char sAlo[2][BLK_M * LDSB];
  __shared__ __align__(16) unsigned char sBhi[2][BLK_N * LDSB];
  __shared__ __align__(16) unsigned char sBlo[2][BLK_N * LDSB];

  const int tid   = threadIdx.x;
  const int lane  = tid & 31;
  const int wave  = tid >> 5;
  const int waveM = wave >> 2;
  const int waveN = wave & 3;
  const int m0 = blockIdx.y * BLK_M;
  const int n0 = blockIdx.x * BLK_N;

  const int ldRow = tid >> 1;
  const int ldCol = (tid & 1) << 4;
  const unsigned sRowOff = (unsigned)ldRow * LDSB + (unsigned)ldCol * 2u;

  const int lr = lane & 15;
  const int lh = (lane >> 4) & 1;

  unsigned aFragOff[4], bFragOff[2];
#pragma unroll
  for (int mi = 0; mi < 4; ++mi)
    aFragOff[mi] =
        (unsigned)(waveM * 64 + mi * 16 + lr) * LDSB + (unsigned)lh * 16u;
#pragma unroll
  for (int ni = 0; ni < 2; ++ni)
    bFragOff[ni] =
        (unsigned)(waveN * 32 + ni * 16 + lr) * LDSB + (unsigned)lh * 16u;

  const v8f vzero = {0.f, 0.f, 0.f, 0.f, 0.f, 0.f, 0.f, 0.f};
  v8f acc[4][2];
#pragma unroll
  for (int mi = 0; mi < 4; ++mi)
#pragma unroll
    for (int ni = 0; ni < 2; ++ni) acc[mi][ni] = vzero;

  auto load_tile = [&](Tile& tl, int t) {
    const int r  = t / KCHUNKS;
    const int k0 = (t - r * KCHUNKS) * BLK_K;
    const float* aPtr =
        A + ((size_t)r * M_DIM + (size_t)(m0 + ldRow)) * (size_t)K_DIM +
        (k0 + ldCol);
    const float* bPtr =
        B + ((size_t)r * N_DIM + (size_t)(n0 + ldRow)) * (size_t)K_DIM +
        (k0 + ldCol);
#pragma unroll
    for (int i = 0; i < 4; ++i) {
      tl.a[i] = *(const f4*)(aPtr + i * 4);
      tl.b[i] = *(const f4*)(bPtr + i * 4);
    }
  };

  auto stage_tile = [&](const Tile& tl, int buf) {
    ui4 ah[2], al[2], bh[2], bl[2];
#pragma unroll
    for (int i = 0; i < 4; ++i) {
#pragma unroll
      for (int j = 0; j < 4; j += 2) {
        const int e = (i * 4 + j) >> 1;
        unsigned h, l;
        cvt_split_pair(tl.a[i][j], tl.a[i][j + 1], h, l);
        ah[e >> 2][e & 3] = h;
        al[e >> 2][e & 3] = l;
        cvt_split_pair(tl.b[i][j], tl.b[i][j + 1], h, l);
        bh[e >> 2][e & 3] = h;
        bl[e >> 2][e & 3] = l;
      }
    }
    *(ui4*)(sAhi[buf] + sRowOff)      = ah[0];
    *(ui4*)(sAhi[buf] + sRowOff + 16) = ah[1];
    *(ui4*)(sAlo[buf] + sRowOff)      = al[0];
    *(ui4*)(sAlo[buf] + sRowOff + 16) = al[1];
    *(ui4*)(sBhi[buf] + sRowOff)      = bh[0];
    *(ui4*)(sBhi[buf] + sRowOff + 16) = bh[1];
    *(ui4*)(sBlo[buf] + sRowOff)      = bl[0];
    *(ui4*)(sBlo[buf] + sRowOff + 16) = bl[1];
  };

  auto compute_tile = [&](int buf) {
    FragBF fahi[4], falo[4];
#pragma unroll
    for (int mi = 0; mi < 4; ++mi) {
      const unsigned off = aFragOff[mi];
      fahi[mi].h[0] = *(const v8bf*)(sAhi[buf] + off);
      fahi[mi].h[1] = *(const v8bf*)(sAhi[buf] + off + 32);
      falo[mi].h[0] = *(const v8bf*)(sAlo[buf] + off);
      falo[mi].h[1] = *(const v8bf*)(sAlo[buf] + off + 32);
    }
    FragBF fbhi[2], fblo[2];
#pragma unroll
    for (int ni = 0; ni < 2; ++ni) {
      const unsigned off = bFragOff[ni];
      fbhi[ni].h[0] = *(const v8bf*)(sBhi[buf] + off);
      fbhi[ni].h[1] = *(const v8bf*)(sBhi[buf] + off + 32);
      fblo[ni].h[0] = *(const v8bf*)(sBlo[buf] + off);
      fblo[ni].h[1] = *(const v8bf*)(sBlo[buf] + off + 32);
    }
#pragma unroll
    for (int mi = 0; mi < 4; ++mi) {
#pragma unroll
      for (int ni = 0; ni < 2; ++ni) {
        acc[mi][ni] = __builtin_amdgcn_wmma_f32_16x16x32_bf16(
            false, fahi[mi].v, false, fbhi[ni].v, (short)0, acc[mi][ni],
            false, false);
        acc[mi][ni] = __builtin_amdgcn_wmma_f32_16x16x32_bf16(
            false, falo[mi].v, false, fbhi[ni].v, (short)0, acc[mi][ni],
            false, false);
        acc[mi][ni] = __builtin_amdgcn_wmma_f32_16x16x32_bf16(
            false, fahi[mi].v, false, fblo[ni].v, (short)0, acc[mi][ni],
            false, false);
      }
    }
  };

  {
    Tile t0;
    load_tile(t0, 0);
    stage_tile(t0, 0);
  }
  __syncthreads();

  for (int t = 0; t < TOTAL_T; ++t) {
    const int pb = t & 1;
    Tile nxt;
    if (t + 1 < TOTAL_T) load_tile(nxt, t + 1);
    compute_tile(pb);
    if (t + 1 < TOTAL_T) stage_tile(nxt, pb ^ 1);
    __syncthreads();
  }

#pragma unroll
  for (int mi = 0; mi < 4; ++mi) {
#pragma unroll
    for (int ni = 0; ni < 2; ++ni) {
      const int row0 = m0 + waveM * 64 + mi * 16 + lh * 8;
      const int col  = n0 + waveN * 32 + ni * 16 + lr;
      float* cp = C + (size_t)row0 * N_DIM + col;
#pragma unroll
      for (int j = 0; j < 8; ++j) cp[(size_t)j * N_DIM] = acc[mi][ni][j];
    }
  }
}

extern "C" void kernel_launch(void* const* d_in, const int* in_sizes, int n_in,
                              void* d_out, int out_size, void* d_ws,
                              size_t ws_size, hipStream_t stream) {
  (void)in_sizes; (void)n_in; (void)out_size;
  const float* A = (const float*)d_in[0];   // [8, 4096, 1792]
  const float* B = (const float*)d_in[1];   // [8, 4096, 1792]
  float*       C = (float*)d_out;           // [4096, 4096]

  dim3 block(256);

  if (ws_size >= 4 * PLANE_BYTES && d_ws != nullptr) {
    unsigned short* Ahi = (unsigned short*)d_ws;
    unsigned short* Alo = Ahi + PLANE_ELEMS;
    unsigned short* Bhi = Alo + PLANE_ELEMS;
    unsigned short* Blo = Bhi + PLANE_ELEMS;
    const unsigned convThreads = 2u * TOTAL_T * 4096u;      // 3,670,016
    convert_planes<<<dim3(convThreads / 256u), block, 0, stream>>>(
        A, B, Ahi, Alo, Bhi, Blo);
    dim3 grid(N_DIM / PBLK_N, M_DIM / PBLK_M);              // 32 x 16
    gemm_planes<<<grid, block, 0, stream>>>(Ahi, Alo, Bhi, Blo, C);
  } else {
    dim3 grid(N_DIM / BLK_N, M_DIM / BLK_M);                // 32 x 32
    gemm_fused<<<grid, block, 0, stream>>>(A, B, C);
  }
}
